// QuantViTSelfAttention_2645699854485
// MI455X (gfx1250) — matrix operand order repaired and checked by reading, not compiled, not run on hardware
//
#include <hip/hip_runtime.h>
#include <stdint.h>

// ---------------------------------------------------------------------------
// Problem constants (from reference): B=16, S=1024, D=1024, H=16, Dh=64
// ---------------------------------------------------------------------------
#define BATCH  16
#define SEQ    1024
#define DMODEL 1024
#define NHEAD  16
#define HDIM   64
#define SOFTMAX_SCALE 0.125f   // 1/sqrt(64)

typedef __attribute__((ext_vector_type(16))) __bf16 v16bf;
typedef __attribute__((ext_vector_type(8)))  float  v8f;

// Native truncation; backend picks v_cvt for gfx1250 (bf16 is a legal type).
__device__ __forceinline__ __bf16 f2bf(float f) { return (__bf16)f; }

// Load a 32-byte bf16 fragment as two 16B chunks.
__device__ __forceinline__ v16bf ldfrag(const __bf16* p0, const __bf16* p1) {
    union { v16bf v; uint4 u[2]; } f;
    f.u[0] = *(const uint4*)p0;
    f.u[1] = *(const uint4*)p1;
    return f.v;
}

// Cross-lane reductions within a 16-lane group (wave32; lanes 0-15 and 16-31
// are independent groups, matching the WMMA C/D fragment row split).
__device__ __forceinline__ float group16_max(float v) {
    v = fmaxf(v, __shfl_xor(v, 1, 32));
    v = fmaxf(v, __shfl_xor(v, 2, 32));
    v = fmaxf(v, __shfl_xor(v, 4, 32));
    v = fmaxf(v, __shfl_xor(v, 8, 32));
    return v;
}
__device__ __forceinline__ float group16_sum(float v) {
    v += __shfl_xor(v, 1, 32);
    v += __shfl_xor(v, 2, 32);
    v += __shfl_xor(v, 4, 32);
    v += __shfl_xor(v, 8, 32);
    return v;
}

// ---------------------------------------------------------------------------
// Kernel 0: bulk fp32 -> bf16 conversion (one-time; removes all conversion
// VALU from the GEMM inner loop and halves its operand traffic).
// ---------------------------------------------------------------------------
__global__ __launch_bounds__(256) void cvt_f32_bf16_kernel(
    const float* __restrict__ in, __bf16* __restrict__ out, int n8)
{
    const int i = blockIdx.x * 256 + threadIdx.x;
    if (i >= n8) return;
    const float4 a = ((const float4*)in)[2 * i + 0];
    const float4 b = ((const float4*)in)[2 * i + 1];
    union { __bf16 t[8]; uint4 u; } pk;
    pk.t[0] = f2bf(a.x); pk.t[1] = f2bf(a.y); pk.t[2] = f2bf(a.z); pk.t[3] = f2bf(a.w);
    pk.t[4] = f2bf(b.x); pk.t[5] = f2bf(b.y); pk.t[6] = f2bf(b.z); pk.t[7] = f2bf(b.w);
    ((uint4*)out)[i] = pk.u;
}

// ---------------------------------------------------------------------------
// Kernel 1: fused QKV GEMM on pre-converted bf16 operands.
//   y = x @ W^T + b   (W rows are contiguous over k -> direct B fragments)
// One wave computes a 16x64 output tile (4 accumulators): the A fragment is
// reused across 4 WMMAs per k-step. A 64-col group == one head (Dh=64).
//   Q, K stored [B, H, S, Dh];  V stored transposed [B, H, Dh, S].
// ---------------------------------------------------------------------------
__global__ __launch_bounds__(128) void qkv_gemm_kernel(
    const __bf16* __restrict__ Xbf,
    const __bf16* __restrict__ Wqb, const __bf16* __restrict__ Wkb,
    const __bf16* __restrict__ Wvb,
    const float* __restrict__ bq, const float* __restrict__ bk,
    const float* __restrict__ bv,
    __bf16* __restrict__ Qws, __bf16* __restrict__ Kws, __bf16* __restrict__ VTws)
{
    const int lane = threadIdx.x & 31;
    const int wave = threadIdx.x >> 5;
    const int ln   = lane & 15;     // A: row within tile; B: col within tile
    const int hi   = lane >> 4;     // lane-group select

    const int m0   = blockIdx.x * 16;                 // output row base
    const int ng   = blockIdx.y * 4 + wave;           // 0..47: 64-col group
    const int n0g  = ng * 64;
    const int mat  = n0g / DMODEL;                    // 0=Q, 1=K, 2=V
    const int col0 = n0g % DMODEL;                    // multiple of 64 -> head base

    const __bf16* W    = (mat == 0) ? Wqb : (mat == 1) ? Wkb : Wvb;
    const float*  bias = (mat == 0) ? bq  : (mat == 1) ? bk  : bv;

    const __bf16* arow = Xbf + (size_t)(m0 + ln) * DMODEL;
    const __bf16* brow[4];
#pragma unroll
    for (int t = 0; t < 4; ++t)
        brow[t] = W + (size_t)(col0 + 16 * t + ln) * DMODEL;

    v8f c[4] = {{}, {}, {}, {}};
    for (int k0 = 0; k0 < DMODEL; k0 += 32) {
        __builtin_prefetch(arow + k0 + 64, 0, 3);     // global_prefetch_b8
        // A fragment (16x32): lane<16 holds K {0..7,16..23}, lane>=16 {8..15,24..31}
        const v16bf a = ldfrag(arow + k0 + 8 * hi, arow + k0 + 16 + 8 * hi);
#pragma unroll
        for (int t = 0; t < 4; ++t) {
            // B fragment (32x16): lane<16 holds K {0..7,16..23}, lane>=16 {8..15,24..31}
            const v16bf b = ldfrag(brow[t] + k0 + 8 * hi,
                                   brow[t] + k0 + 16 + 8 * hi);
            c[t] = __builtin_amdgcn_wmma_f32_16x16x32_bf16(
                       false, a, false, b, (short)0, c[t], false, false);
        }
    }

    // Epilogue: bias add, convert, store.
    const int h    = col0 / HDIM;
    const int bidx = m0 / SEQ;
    const int s0   = m0 % SEQ;

#pragma unroll
    for (int t = 0; t < 4; ++t) {
        const int   d  = 16 * t + ln;                 // 0..63 within the head
        const float cb = bias[col0 + d];
        if (mat < 2) {
            __bf16* dst = ((mat == 0) ? Qws : Kws)
                        + ((size_t)(bidx * NHEAD + h) * SEQ) * HDIM;
#pragma unroll
            for (int r = 0; r < 8; ++r) {
                const int srow = s0 + hi * 8 + r;     // lanes>=16 hold rows 8..15
                dst[(size_t)srow * HDIM + d] = f2bf(c[t][r] + cb);
            }
        } else {
            // V transposed: [b,h,d,s]; 8 values contiguous along s -> 16B store
            union { __bf16 q[8]; uint4 u; } pk;
#pragma unroll
            for (int r = 0; r < 8; ++r) pk.q[r] = f2bf(c[t][r] + cb);
            const size_t base =
                ((size_t)(bidx * NHEAD + h) * HDIM + d) * SEQ + s0 + hi * 8;
            *(uint4*)(VTws + base) = pk.u;
        }
    }
}

// ---------------------------------------------------------------------------
// Kernel 2: flash attention. One wave per (b, h, 32-row q-tile): two 16-row
// groups g=0,1 share every K/V B-fragment load (2 WMMAs per fragment),
// halving L2 traffic vs a 16-row tile. Online softmax over 32-key chunks;
// P transposed C->A layout through LDS. 512 wmma per wave.
// ---------------------------------------------------------------------------
__global__ __launch_bounds__(256) void flash_attn_kernel(
    const __bf16* __restrict__ Qws, const __bf16* __restrict__ Kws,
    const __bf16* __restrict__ VTws, float* __restrict__ out)
{
    __shared__ __bf16 ldsP[8][2][16 * 32];   // per-wave, per-group 16x32 P tile

    const int lane = threadIdx.x & 31;
    const int wave = threadIdx.x >> 5;
    const int ln   = lane & 15;
    const int hi   = lane >> 4;

    const int gw = blockIdx.x * 8 + wave;        // 0 .. B*H*(S/32)-1
    const int qt = gw & (SEQ / 32 - 1);          // 32-row q-tile
    const int h  = (gw >> 5) & (NHEAD - 1);
    const int b  = gw >> 9;
    const int q0 = qt * 32;

    const __bf16* Qbase = Qws  + ((size_t)(b * NHEAD + h) * SEQ) * HDIM;
    const __bf16* Kbase = Kws  + ((size_t)(b * NHEAD + h) * SEQ) * HDIM;
    const __bf16* Vbase = VTws + ((size_t)(b * NHEAD + h) * HDIM) * SEQ;

    // Q A-fragments: group g rows q0+16g.., frag dt covers d = dt*32..dt*32+31.
    v16bf qA[2][2];
#pragma unroll
    for (int g = 0; g < 2; ++g) {
        const __bf16* qrow = Qbase + (size_t)(q0 + g * 16 + ln) * HDIM;
#pragma unroll
        for (int dt = 0; dt < 2; ++dt)
            qA[g][dt] = ldfrag(qrow + dt * 32 + 8 * hi,
                               qrow + dt * 32 + 16 + 8 * hi);
    }

    v8f acc[2][4] = {{{}, {}, {}, {}}, {{}, {}, {}, {}}};
    float mrun[2][8], lrun[2][8];
#pragma unroll
    for (int g = 0; g < 2; ++g)
#pragma unroll
        for (int r = 0; r < 8; ++r) { mrun[g][r] = -INFINITY; lrun[g][r] = 0.0f; }

    for (int j0 = 0; j0 < SEQ; j0 += 32) {
        // ---- scores: per group, two 16x16 tiles; K frags shared by groups ----
        v8f sf[2][2] = {{{}, {}}, {{}, {}}};
#pragma unroll
        for (int kt = 0; kt < 2; ++kt) {
            const int key0 = j0 + kt * 16;
#pragma unroll
            for (int dt = 0; dt < 2; ++dt) {
                // B frag: lane col = key0+ln, K values = d (two 16B chunks)
                const __bf16* p =
                    Kbase + (size_t)(key0 + ln) * HDIM + dt * 32 + 8 * hi;
                const v16bf kb = ldfrag(p, p + 16);
#pragma unroll
                for (int g = 0; g < 2; ++g)
                    sf[g][kt] = __builtin_amdgcn_wmma_f32_16x16x32_bf16(
                                    false, qA[g][dt], false, kb,
                                    (short)0, sf[g][kt], false, false);
            }
        }

        // ---- online softmax over these 32 keys, per group ----
#pragma unroll
        for (int g = 0; g < 2; ++g) {
            float alpha[8];
#pragma unroll
            for (int r = 0; r < 8; ++r) {
                const float sa = sf[g][0][r] * SOFTMAX_SCALE;
                const float sb = sf[g][1][r] * SOFTMAX_SCALE;
                const float tm = group16_max(fmaxf(sa, sb));
                const float nm = fmaxf(mrun[g][r], tm);
                alpha[r] = __expf(mrun[g][r] - nm);
                const float pa = __expf(sa - nm);
                const float pb = __expf(sb - nm);
                lrun[g][r] = lrun[g][r] * alpha[r] + group16_sum(pa + pb);
                mrun[g][r] = nm;
                sf[g][0][r] = pa;
                sf[g][1][r] = pb;
            }
#pragma unroll
            for (int t = 0; t < 4; ++t)
#pragma unroll
                for (int r = 0; r < 8; ++r) acc[g][t][r] *= alpha[r];

            // transpose P (C/D layout) -> A layout via LDS
            __bf16* P = ldsP[wave][g];
#pragma unroll
            for (int r = 0; r < 8; ++r) {
                const int row = hi * 8 + r;
                P[row * 32 + ln]      = f2bf(sf[g][0][r]);
                P[row * 32 + 16 + ln] = f2bf(sf[g][1][r]);
            }
        }
        asm volatile("s_wait_dscnt 0" ::: "memory");  // wave-internal DS ordering

        v16bf pA[2];
#pragma unroll
        for (int g = 0; g < 2; ++g) {
            const __bf16* pr = ldsP[wave][g] + ln * 32;
            pA[g] = ldfrag(pr + 8 * hi, pr + 16 + 8 * hi);
        }

        // ---- ctx += P * V; V frags shared by both groups ----
#pragma unroll
        for (int t = 0; t < 4; ++t) {
            const __bf16* p = Vbase + (size_t)(t * 16 + ln) * SEQ + j0 + 8 * hi;
            const v16bf vb = ldfrag(p, p + 16);
#pragma unroll
            for (int g = 0; g < 2; ++g)
                acc[g][t] = __builtin_amdgcn_wmma_f32_16x16x32_bf16(
                                false, pA[g], false, vb,
                                (short)0, acc[g][t], false, false);
        }
    }

    // ---- epilogue: normalize and write [B, S, H*Dh] fp32 ----
#pragma unroll
    for (int g = 0; g < 2; ++g) {
        float inv[8];
#pragma unroll
        for (int r = 0; r < 8; ++r) inv[r] = 1.0f / lrun[g][r];
#pragma unroll
        for (int t = 0; t < 4; ++t) {
            const int col = h * HDIM + t * 16 + ln;
#pragma unroll
            for (int r = 0; r < 8; ++r) {
                const int srow = q0 + g * 16 + hi * 8 + r;
                out[((size_t)b * SEQ + srow) * DMODEL + col] = acc[g][t][r] * inv[r];
            }
        }
    }
}

// ---------------------------------------------------------------------------
// Launch. d_ws layout (bf16 elements):
//   Q [16M] | K [16M] | V^T [16M] | Xbf [16M] | Wq [1M] | Wk [1M] | Wv [1M]
//   = 67M bf16 = 134 MB of scratch.
// ---------------------------------------------------------------------------
extern "C" void kernel_launch(void* const* d_in, const int* in_sizes, int n_in,
                              void* d_out, int out_size, void* d_ws, size_t ws_size,
                              hipStream_t stream) {
    (void)in_sizes; (void)n_in; (void)out_size; (void)ws_size;
    const float* X  = (const float*)d_in[0];
    const float* Wq = (const float*)d_in[1];
    const float* bq = (const float*)d_in[2];
    const float* Wk = (const float*)d_in[3];
    const float* bk = (const float*)d_in[4];
    const float* Wv = (const float*)d_in[5];
    const float* bv = (const float*)d_in[6];
    float* out = (float*)d_out;

    const size_t perQKV = (size_t)BATCH * NHEAD * SEQ * HDIM;   // 16M elements
    const size_t perW   = (size_t)DMODEL * DMODEL;              // 1M elements
    __bf16* Qws  = (__bf16*)d_ws;
    __bf16* Kws  = Qws + perQKV;
    __bf16* VTws = Kws + perQKV;
    __bf16* Xbf  = VTws + perQKV;
    __bf16* Wqb  = Xbf + (size_t)BATCH * SEQ * DMODEL;
    __bf16* Wkb  = Wqb + perW;
    __bf16* Wvb  = Wkb + perW;

    // Kernel 0: one-time fp32 -> bf16 conversions
    const int nX8 = (BATCH * SEQ * DMODEL) / 8;
    const int nW8 = (int)(perW / 8);
    cvt_f32_bf16_kernel<<<nX8 / 256, 256, 0, stream>>>(X, Xbf, nX8);
    cvt_f32_bf16_kernel<<<nW8 / 256, 256, 0, stream>>>(Wq, Wqb, nW8);
    cvt_f32_bf16_kernel<<<nW8 / 256, 256, 0, stream>>>(Wk, Wkb, nW8);
    cvt_f32_bf16_kernel<<<nW8 / 256, 256, 0, stream>>>(Wv, Wvb, nW8);

    // Kernel 1: (B*S/16) m-tiles x 12 blocks of 4 waves (each wave 16x64)
    dim3 g1(BATCH * SEQ / 16, (3 * DMODEL / 64) / 4);
    qkv_gemm_kernel<<<g1, 128, 0, stream>>>(Xbf, Wqb, Wkb, Wvb, bq, bk, bv,
                                            Qws, Kws, VTws);

    // Kernel 2: B*H*(S/32) waves (32 q-rows each), 8 waves per block
    const int nwaves = BATCH * NHEAD * (SEQ / 32);
    flash_attn_kernel<<<nwaves / 8, 256, 0, stream>>>(Qws, Kws, VTws, out);
}
